// JoinABLe_18442589570219
// MI455X (gfx1250) — matrix operand order repaired and checked by reading, not compiled, not run on hardware
//
#include <hip/hip_runtime.h>
#include <hip/hip_bf16.h>

// ---------------------------------------------------------------------------
// JoinABLe GNN forward on gfx1250 (MI455X), bf16 WMMA GEMM core.
// GEMM: 16x64 output per wave, 4 WMMAs / K-step, double-buffered K pipeline.
// ---------------------------------------------------------------------------

#define NN      20000      // nodes per graph
#define DIMC    256        // hidden dim
#define DFP     704        // face feat padded (700 -> 704, mult of 32)
#define DEP     96         // edge feat padded (70 -> 96)
#define E1C     640000     // edges per part graph
#define ETC     (E1C + NN) // edges + self loops
#define EJC     400000     // joint candidate pairs
#define LRELU   0.2f
#define BN_EPS  1e-5f

typedef __attribute__((ext_vector_type(16))) __bf16         v16bf;
typedef __attribute__((ext_vector_type(8)))  float          v8f;
typedef __attribute__((ext_vector_type(8)))  unsigned short ush8;

union BF16Tile { ush8 h[2]; v16bf v; };

// f32 -> bf16 (RNE) as raw u16
static __device__ __forceinline__ unsigned short f2bfu(float f) {
    unsigned u = __float_as_uint(f);
    unsigned r = u + 0x7FFFu + ((u >> 16) & 1u);
    return (unsigned short)(r >> 16);
}
// order-preserving float<->uint key for atomic max on signed floats
static __device__ __forceinline__ unsigned fkey(float f) {
    unsigned u = __float_as_uint(f);
    return (u & 0x80000000u) ? ~u : (u | 0x80000000u);
}
static __device__ __forceinline__ float fdec(unsigned k) {
    unsigned u = (k & 0x80000000u) ? (k & 0x7FFFFFFFu) : ~k;
    return __uint_as_float(u);
}

// A-fragment: 16x32 bf16 tile, per-lane K striping (lane-half selects K sub-block)
static __device__ __forceinline__ v16bf ldA(const unsigned short* row, int k0, int hl) {
    BF16Tile t;
    t.h[0] = *(const ush8*)(row + k0 + 8 * hl);
    t.h[1] = *(const ush8*)(row + k0 + 16 + 8 * hl);
    return t.v;
}
// B-fragment: 32x16 bf16 tile from N-major transposed weights
static __device__ __forceinline__ v16bf ldB(const unsigned short* row, int k0, int hl) {
    BF16Tile t;
    t.h[0] = *(const ush8*)(row + k0 + 8 * hl);
    t.h[1] = *(const ush8*)(row + k0 + 16 + 8 * hl);
    return t.v;
}
static __device__ __forceinline__ v8f wmma_bf16(v16bf a, v16bf b, v8f c) {
    return __builtin_amdgcn_wmma_f32_16x16x32_bf16(false, a, false, b, (short)0, c, false, false);
}

// ---------------------------------------------------------------------------
// Conversion / init kernels
// ---------------------------------------------------------------------------

// f32 [M][K] -> bf16 [M][Kp], zero-padding K..Kp-1
__global__ void k_cvt_pad(const float* __restrict__ src, unsigned short* __restrict__ dst,
                          int M, int K, int Kp) {
    long g = (long)blockIdx.x * blockDim.x + threadIdx.x;
    long tot = (long)M * Kp;
    if (g >= tot) return;
    int m = (int)(g / Kp), k = (int)(g % Kp);
    dst[g] = (k < K) ? f2bfu(src[(size_t)m * K + k]) : (unsigned short)0;
}

// f32 weight [K][256] -> bf16 transposed dst[n][koff + k], k in [0,Kspan), zero pad k>=K
__global__ void k_cvt_wT(const float* __restrict__ src, unsigned short* __restrict__ dst,
                         int K, int Kspan, int koff, int Kp) {
    long g = (long)blockIdx.x * blockDim.x + threadIdx.x;
    long tot = (long)DIMC * Kspan;
    if (g >= tot) return;
    int n = (int)(g / Kspan), k = (int)(g % Kspan);
    dst[(size_t)n * Kp + koff + k] = (k < K) ? f2bfu(src[(size_t)k * DIMC + n]) : (unsigned short)0;
}

__global__ void k_fill_u32(unsigned* p, unsigned v, int n) {
    int g = blockIdx.x * blockDim.x + threadIdx.x;
    if (g < n) p[g] = v;
}

// agg[n][c] = bias[c]
__global__ void k_init_rows(float* __restrict__ dst, const float* __restrict__ bias, int rows) {
    long g = (long)blockIdx.x * blockDim.x + threadIdx.x;
    if (g >= (long)rows * DIMC) return;
    dst[g] = bias[g & (DIMC - 1)];
}

__global__ void k_init_scalar(float* dst, const float* s, int n) {
    int g = blockIdx.x * blockDim.x + threadIdx.x;
    if (g < n) dst[g] = s[0];
}

__global__ void k_vec_add(const float* a, const float* b, float* c, int n) {
    int g = blockIdx.x * blockDim.x + threadIdx.x;
    if (g < n) c[g] = a[g] + b[g];
}

__global__ void k_cvt_bf16(const float* __restrict__ src, unsigned short* __restrict__ dst, long n) {
    long g = (long)blockIdx.x * blockDim.x + threadIdx.x;
    if (g < n) dst[g] = f2bfu(src[g]);
}

// ---------------------------------------------------------------------------
// Core bf16 WMMA GEMM: C[M][256] = epi(A[M][K] @ BT[256][K]^T + bias)
//   block = 128 threads = 4 waves; wave w computes a 16x64 tile
//   (rows 16*bx .. +16, cols 64*w .. +64) via 4 accumulators; the K loop is
//   double-buffered so each s_wait covers loads issued one iteration earlier.
//   epi: 0 = relu -> bf16 out, 1 = f32 out, 2 = bf16 out (no relu)
// ---------------------------------------------------------------------------
__global__ void k_gemm_bf16(const unsigned short* __restrict__ A,
                            const unsigned short* __restrict__ BT,
                            int K, const float* __restrict__ bias, int epi,
                            float* __restrict__ outF, unsigned short* __restrict__ outB,
                            int ldo) {
    const int lane = threadIdx.x & 31;
    const int wave = threadIdx.x >> 5;
    const int hl = lane >> 4, lm = lane & 15;
    const int m0 = blockIdx.x * 16;
    const int nb = wave * 64;
    const unsigned short* a_row = A + (size_t)(m0 + lm) * K;
    const unsigned short* b_row[4];
#pragma unroll
    for (int j = 0; j < 4; ++j) b_row[j] = BT + (size_t)(nb + j * 16 + lm) * K;

    v8f acc[4] = {{}, {}, {}, {}};
    v16bf a = ldA(a_row, 0, hl);
    v16bf b[4];
#pragma unroll
    for (int j = 0; j < 4; ++j) b[j] = ldB(b_row[j], 0, hl);

    for (int k0 = 32; k0 < K; k0 += 32) {
        __builtin_prefetch(a_row + k0 + 32, 0, 1);
        v16bf an = ldA(a_row, k0, hl);
        v16bf bn[4];
#pragma unroll
        for (int j = 0; j < 4; ++j) bn[j] = ldB(b_row[j], k0, hl);
#pragma unroll
        for (int j = 0; j < 4; ++j) acc[j] = wmma_bf16(a, b[j], acc[j]);
        a = an;
#pragma unroll
        for (int j = 0; j < 4; ++j) b[j] = bn[j];
    }
#pragma unroll
    for (int j = 0; j < 4; ++j) acc[j] = wmma_bf16(a, b[j], acc[j]);

#pragma unroll
    for (int j = 0; j < 4; ++j) {
        const int n = nb + j * 16 + lm;
        const float bv = bias[n];
#pragma unroll
        for (int r = 0; r < 8; ++r) {
            int m = m0 + r + 8 * hl;
            float v = acc[j][r] + bv;
            if (epi == 0)      { v = v > 0.f ? v : 0.f; outB[(size_t)m * ldo + n] = f2bfu(v); }
            else if (epi == 1) { outF[(size_t)m * ldo + n] = v; }
            else               { outB[(size_t)m * ldo + n] = f2bfu(v); }
        }
    }
}

// Post layer 1: fused gather GEMM. Row m of A = concat(X1[i0[m]], X2[i1[m]]) (K=512).
__global__ void k_gemm_gather(const unsigned short* __restrict__ X1,
                              const unsigned short* __restrict__ X2,
                              const int* __restrict__ i0, const int* __restrict__ i1,
                              const unsigned short* __restrict__ BT,
                              const float* __restrict__ bias,
                              unsigned short* __restrict__ outB) {
    const int lane = threadIdx.x & 31;
    const int wave = threadIdx.x >> 5;
    const int hl = lane >> 4, lm = lane & 15;
    const int m0 = blockIdx.x * 16;
    const int nb = wave * 64;
    const int m = m0 + lm;
    const unsigned short* a1 = X1 + (size_t)i0[m] * DIMC;
    const unsigned short* a2 = X2 + (size_t)i1[m] * DIMC;
    const unsigned short* b_row[4];
#pragma unroll
    for (int j = 0; j < 4; ++j) b_row[j] = BT + (size_t)(nb + j * 16 + lm) * 512;

    v8f acc[4] = {{}, {}, {}, {}};
    v16bf a = ldA(a1, 0, hl);
    v16bf b[4];
#pragma unroll
    for (int j = 0; j < 4; ++j) b[j] = ldB(b_row[j], 0, hl);

    for (int k0 = 32; k0 < 512; k0 += 32) {
        const unsigned short* abase = (k0 < DIMC) ? (a1 + k0) : (a2 + (k0 - DIMC));
        v16bf an = ldA(abase, 0, hl);
        v16bf bn[4];
#pragma unroll
        for (int j = 0; j < 4; ++j) bn[j] = ldB(b_row[j], k0, hl);
#pragma unroll
        for (int j = 0; j < 4; ++j) acc[j] = wmma_bf16(a, b[j], acc[j]);
        a = an;
#pragma unroll
        for (int j = 0; j < 4; ++j) b[j] = bn[j];
    }
#pragma unroll
    for (int j = 0; j < 4; ++j) acc[j] = wmma_bf16(a, b[j], acc[j]);

#pragma unroll
    for (int j = 0; j < 4; ++j) {
        const int n = nb + j * 16 + lm;
        const float bv = bias[n];
#pragma unroll
        for (int r = 0; r < 8; ++r) {
            float v = acc[j][r] + bv;
            v = v > 0.f ? v : 0.f;
            outB[(size_t)(m0 + r + 8 * hl) * DIMC + n] = f2bfu(v);
        }
    }
}

// Post layers 2+3 fused: out[m] += sum_n relu(A@W2 + b2)[m][n] * w3[n]  (K=256)
__global__ void k_gemm_post2(const unsigned short* __restrict__ A,
                             const unsigned short* __restrict__ BT,
                             const float* __restrict__ b2, const float* __restrict__ w3,
                             float* __restrict__ out) {
    const int lane = threadIdx.x & 31;
    const int wave = threadIdx.x >> 5;
    const int hl = lane >> 4, lm = lane & 15;
    const int m0 = blockIdx.x * 16;
    const int nb = wave * 64;
    const unsigned short* a_row = A + (size_t)(m0 + lm) * DIMC;
    const unsigned short* b_row[4];
#pragma unroll
    for (int j = 0; j < 4; ++j) b_row[j] = BT + (size_t)(nb + j * 16 + lm) * DIMC;

    v8f acc[4] = {{}, {}, {}, {}};
    v16bf a = ldA(a_row, 0, hl);
    v16bf b[4];
#pragma unroll
    for (int j = 0; j < 4; ++j) b[j] = ldB(b_row[j], 0, hl);

    for (int k0 = 32; k0 < DIMC; k0 += 32) {
        __builtin_prefetch(a_row + k0 + 32, 0, 1);
        v16bf an = ldA(a_row, k0, hl);
        v16bf bn[4];
#pragma unroll
        for (int j = 0; j < 4; ++j) bn[j] = ldB(b_row[j], k0, hl);
#pragma unroll
        for (int j = 0; j < 4; ++j) acc[j] = wmma_bf16(a, b[j], acc[j]);
        a = an;
#pragma unroll
        for (int j = 0; j < 4; ++j) b[j] = bn[j];
    }
#pragma unroll
    for (int j = 0; j < 4; ++j) acc[j] = wmma_bf16(a, b[j], acc[j]);

    float part[8] = {0, 0, 0, 0, 0, 0, 0, 0};
#pragma unroll
    for (int j = 0; j < 4; ++j) {
        const int n = nb + j * 16 + lm;
        const float bv = b2[n], wn = w3[n];
#pragma unroll
        for (int r = 0; r < 8; ++r) {
            float v = acc[j][r] + bv;
            v = v > 0.f ? v : 0.f;
            part[r] += v * wn;          // accumulate 4 n-tiles before atomics
        }
    }
#pragma unroll
    for (int r = 0; r < 8; ++r)
        atomicAdd(&out[m0 + r + 8 * hl], part[r]);
}

// ---------------------------------------------------------------------------
// GATv2 edge kernels (f32). Edges e<E1 from ei, e>=E1 are self loops.
// ---------------------------------------------------------------------------
__global__ void k_gat_logits(const float* __restrict__ xl, const float* __restrict__ xr,
                             const int* __restrict__ ei, const float* __restrict__ att,
                             float* __restrict__ logits, unsigned* __restrict__ maxb) {
    int e = blockIdx.x * blockDim.x + threadIdx.x;
    if (e >= ETC) return;
    int src = (e < E1C) ? ei[e]        : (e - E1C);
    int dst = (e < E1C) ? ei[E1C + e]  : (e - E1C);
    const float4* pl = (const float4*)(xl + (size_t)src * DIMC);
    const float4* pr = (const float4*)(xr + (size_t)dst * DIMC);
    const float4* pa = (const float4*)att;
    float acc[8] = {0, 0, 0, 0, 0, 0, 0, 0};
#pragma unroll 8
    for (int c4 = 0; c4 < 64; ++c4) {
        float4 a = pl[c4], b = pr[c4], w = pa[c4];
        float m;
        m = a.x + b.x; m = m > 0.f ? m : LRELU * m; acc[c4 >> 3] += m * w.x;
        m = a.y + b.y; m = m > 0.f ? m : LRELU * m; acc[c4 >> 3] += m * w.y;
        m = a.z + b.z; m = m > 0.f ? m : LRELU * m; acc[c4 >> 3] += m * w.z;
        m = a.w + b.w; m = m > 0.f ? m : LRELU * m; acc[c4 >> 3] += m * w.w;
    }
#pragma unroll
    for (int h = 0; h < 8; ++h) {
        logits[(size_t)e * 8 + h] = acc[h];
        atomicMax(&maxb[dst * 8 + h], fkey(acc[h]));
    }
}

__global__ void k_gat_softmax(const int* __restrict__ ei, float* __restrict__ logits,
                              const unsigned* __restrict__ maxb, float* __restrict__ den) {
    int g = blockIdx.x * blockDim.x + threadIdx.x;
    if (g >= ETC * 8) return;
    int e = g >> 3, h = g & 7;
    int dst = (e < E1C) ? ei[E1C + e] : (e - E1C);
    float ex = __expf(logits[g] - fdec(maxb[dst * 8 + h]));
    logits[g] = ex;
    atomicAdd(&den[dst * 8 + h], ex);
}

__global__ void k_gat_aggregate(const float* __restrict__ xl, const int* __restrict__ ei,
                                const float* __restrict__ exv, const float* __restrict__ den,
                                float* __restrict__ agg) {
    long g = (long)blockIdx.x * blockDim.x + threadIdx.x;
    if (g >= (long)ETC * 64) return;
    int e = (int)(g >> 6);
    int q = ((int)g & 63) * 4;          // channel base, 4 channels per thread
    int h = q >> 5;
    int src = (e < E1C) ? ei[e]       : (e - E1C);
    int dst = (e < E1C) ? ei[E1C + e] : (e - E1C);
    float alpha = exv[(size_t)e * 8 + h] / den[dst * 8 + h];
    float4 a = *(const float4*)(xl + (size_t)src * DIMC + q);
    float* pd = agg + (size_t)dst * DIMC + q;
    atomicAdd(pd + 0, alpha * a.x);
    atomicAdd(pd + 1, alpha * a.y);
    atomicAdd(pd + 2, alpha * a.z);
    atomicAdd(pd + 3, alpha * a.w);
}

// ---------------------------------------------------------------------------
// BatchNorm (biased var, matching x.var(0)) + ELU + cvt to bf16
// ---------------------------------------------------------------------------
__global__ void k_bn_stats(const float* __restrict__ x, float* __restrict__ mean,
                           float* __restrict__ rstd) {
    __shared__ float sh[256], sh2[256];
    int c = blockIdx.x, tid = threadIdx.x;
    float s = 0.f, s2 = 0.f;
    for (int n = tid; n < NN; n += 256) {
        float v = x[(size_t)n * DIMC + c];
        s += v; s2 += v * v;
    }
    sh[tid] = s; sh2[tid] = s2;
    __syncthreads();
    for (int st = 128; st > 0; st >>= 1) {
        if (tid < st) { sh[tid] += sh[tid + st]; sh2[tid] += sh2[tid + st]; }
        __syncthreads();
    }
    if (tid == 0) {
        float mu = sh[0] / NN;
        float var = sh2[0] / NN - mu * mu;
        mean[c] = mu;
        rstd[c] = rsqrtf(var + BN_EPS);
    }
}

__global__ void k_bn_elu_cvt(const float* __restrict__ x, const float* __restrict__ gamma,
                             const float* __restrict__ beta, const float* __restrict__ mean,
                             const float* __restrict__ rstd, unsigned short* __restrict__ out) {
    long g = (long)blockIdx.x * blockDim.x + threadIdx.x;
    if (g >= (long)NN * DIMC) return;
    int c = (int)(g & (DIMC - 1));
    float v = gamma[c] * (x[g] - mean[c]) * rstd[c] + beta[c];
    v = v > 0.f ? v : (__expf(v) - 1.f);
    out[g] = f2bfu(v);
}

// ---------------------------------------------------------------------------
// Host orchestration
// ---------------------------------------------------------------------------
static inline int cdiv(long a, long b) { return (int)((a + b - 1) / b); }

extern "C" void kernel_launch(void* const* d_in, const int* in_sizes, int n_in,
                              void* d_out, int out_size, void* d_ws, size_t ws_size,
                              hipStream_t stream) {
    (void)in_sizes; (void)n_in; (void)out_size; (void)ws_size;

    // ---- inputs ----
    const float* g1_face = (const float*)d_in[0];
    const float* g1_edge = (const float*)d_in[1];
    const float* g2_face = (const float*)d_in[2];
    const float* g2_edge = (const float*)d_in[3];
    const int*   g1_ei   = (const int*)d_in[4];
    const int*   g2_ei   = (const int*)d_in[5];
    const int*   jg_ei   = (const int*)d_in[6];
    // params (dict insertion order)
    const float* face_W1 = (const float*)d_in[7];
    const float* face_b1 = (const float*)d_in[8];
    const float* face_W2 = (const float*)d_in[9];
    const float* face_b2 = (const float*)d_in[10];
    const float* edge_W1 = (const float*)d_in[11];
    const float* edge_b1 = (const float*)d_in[12];
    const float* edge_W2 = (const float*)d_in[13];
    const float* edge_b2 = (const float*)d_in[14];
    const float* gW[2][2] = {{(const float*)d_in[15], (const float*)d_in[17]},
                             {(const float*)d_in[21], (const float*)d_in[23]}};
    const float* gb[2][2] = {{(const float*)d_in[16], (const float*)d_in[18]},
                             {(const float*)d_in[22], (const float*)d_in[24]}};
    const float* gatt[2]  = {(const float*)d_in[19], (const float*)d_in[25]};
    const float* gbias[2] = {(const float*)d_in[20], (const float*)d_in[26]};
    const float* bn_gamma = (const float*)d_in[27];
    const float* bn_beta  = (const float*)d_in[28];
    const float* post_W1  = (const float*)d_in[29];
    const float* post_b1  = (const float*)d_in[30];
    const float* post_W2  = (const float*)d_in[31];
    const float* post_b2  = (const float*)d_in[32];
    const float* post_W3  = (const float*)d_in[33];
    const float* post_b3  = (const float*)d_in[34];
    float* out = (float*)d_out;

    // ---- workspace bump allocator ----
    char* base = (char*)d_ws;
    size_t off = 0;
    auto alloc = [&](size_t bytes) -> void* {
        void* p = base + off;
        off = (off + bytes + 255) & ~(size_t)255;
        return p;
    };
    typedef unsigned short u16;
    u16* WTf1  = (u16*)alloc((size_t)DIMC * DFP * 2);
    u16* WTe1  = (u16*)alloc((size_t)DIMC * DEP * 2);
    u16* WTcat = (u16*)alloc((size_t)DIMC * 512 * 2);
    u16* WTg[2][2];
    for (int L = 0; L < 2; ++L)
        for (int s = 0; s < 2; ++s) WTg[L][s] = (u16*)alloc((size_t)DIMC * DIMC * 2);
    u16* WTp1 = (u16*)alloc((size_t)DIMC * 512 * 2);
    u16* WTp2 = (u16*)alloc((size_t)DIMC * DIMC * 2);
    u16* Af[2] = {(u16*)alloc((size_t)NN * DFP * 2), (u16*)alloc((size_t)NN * DFP * 2)};
    u16* Ae[2] = {(u16*)alloc((size_t)NN * DEP * 2), (u16*)alloc((size_t)NN * DEP * 2)};
    u16* Hc[2] = {(u16*)alloc((size_t)NN * 512 * 2), (u16*)alloc((size_t)NN * 512 * 2)};
    u16* xbp[2] = {(u16*)alloc((size_t)NN * DIMC * 2), (u16*)alloc((size_t)NN * DIMC * 2)};
    u16* xbm[2] = {(u16*)alloc((size_t)NN * DIMC * 2), (u16*)alloc((size_t)NN * DIMC * 2)};
    u16* xfin[2] = {(u16*)alloc((size_t)NN * DIMC * 2), (u16*)alloc((size_t)NN * DIMC * 2)};
    u16* h1post = (u16*)alloc((size_t)EJC * DIMC * 2);
    float* xl     = (float*)alloc((size_t)NN * DIMC * 4);
    float* xr     = (float*)alloc((size_t)NN * DIMC * 4);
    float* logits = (float*)alloc((size_t)ETC * 8 * 4);
    float* den    = (float*)alloc((size_t)NN * 8 * 4);
    float* agg    = (float*)alloc((size_t)NN * DIMC * 4);
    float* meanb  = (float*)alloc(DIMC * 4);
    float* rstdb  = (float*)alloc(DIMC * 4);
    float* b2sum  = (float*)alloc(DIMC * 4);
    unsigned* maxb = (unsigned*)alloc((size_t)NN * 8 * 4);

    const int T = 256;
    // ---- weight conversion / transpose (f32 -> bf16, padded K, N-major) ----
    k_cvt_wT<<<cdiv((long)DIMC * DFP, T), T, 0, stream>>>(face_W1, WTf1, 700, DFP, 0, DFP);
    k_cvt_wT<<<cdiv((long)DIMC * DEP, T), T, 0, stream>>>(edge_W1, WTe1, 70, DEP, 0, DEP);
    k_cvt_wT<<<cdiv((long)DIMC * 256, T), T, 0, stream>>>(face_W2, WTcat, 256, 256, 0, 512);
    k_cvt_wT<<<cdiv((long)DIMC * 256, T), T, 0, stream>>>(edge_W2, WTcat, 256, 256, 256, 512);
    for (int L = 0; L < 2; ++L)
        for (int s = 0; s < 2; ++s)
            k_cvt_wT<<<cdiv((long)DIMC * 256, T), T, 0, stream>>>(gW[L][s], WTg[L][s], 256, 256, 0, 256);
    k_cvt_wT<<<cdiv((long)DIMC * 512, T), T, 0, stream>>>(post_W1, WTp1, 512, 512, 0, 512);
    k_cvt_wT<<<cdiv((long)DIMC * 256, T), T, 0, stream>>>(post_W2, WTp2, 256, 256, 0, 256);
    k_vec_add<<<1, 256, 0, stream>>>(face_b2, edge_b2, b2sum, 256);

    // ---- input conversion (padded bf16) ----
    k_cvt_pad<<<cdiv((long)NN * DFP, T), T, 0, stream>>>(g1_face, Af[0], NN, 700, DFP);
    k_cvt_pad<<<cdiv((long)NN * DFP, T), T, 0, stream>>>(g2_face, Af[1], NN, 700, DFP);
    k_cvt_pad<<<cdiv((long)NN * DEP, T), T, 0, stream>>>(g1_edge, Ae[0], NN, 70, DEP);
    k_cvt_pad<<<cdiv((long)NN * DEP, T), T, 0, stream>>>(g2_edge, Ae[1], NN, 70, DEP);

    dim3 gN(NN / 16), bG(128);
    const int* ei[2] = {g1_ei, g2_ei};

    // ---- preprocessor: x = face_mlp + edge_mlp (fused via K-concat) ----
    for (int g = 0; g < 2; ++g) {
        k_gemm_bf16<<<gN, bG, 0, stream>>>(Af[g], WTf1, DFP, face_b1, 0, nullptr, Hc[g], 512);
        k_gemm_bf16<<<gN, bG, 0, stream>>>(Ae[g], WTe1, DEP, edge_b1, 0, nullptr, Hc[g] + 256, 512);
        k_gemm_bf16<<<gN, bG, 0, stream>>>(Hc[g], WTcat, 512, b2sum, 2, nullptr, xbp[g], 256);
    }

    // ---- GNN: two GATv2 layers per graph (+ BN/ELU between) ----
    for (int g = 0; g < 2; ++g) {
        for (int L = 0; L < 2; ++L) {
            const u16* xin = (L == 0) ? xbp[g] : xbm[g];
            k_gemm_bf16<<<gN, bG, 0, stream>>>(xin, WTg[L][0], 256, gb[L][0], 1, xl, nullptr, 256);
            k_gemm_bf16<<<gN, bG, 0, stream>>>(xin, WTg[L][1], 256, gb[L][1], 1, xr, nullptr, 256);
            k_fill_u32<<<cdiv(NN * 8, T), T, 0, stream>>>(maxb, 0u, NN * 8);
            k_fill_u32<<<cdiv(NN * 8, T), T, 0, stream>>>((unsigned*)den, 0u, NN * 8);
            k_init_rows<<<cdiv((long)NN * DIMC, T), T, 0, stream>>>(agg, gbias[L], NN);
            k_gat_logits<<<cdiv(ETC, T), T, 0, stream>>>(xl, xr, ei[g], gatt[L], logits, maxb);
            k_gat_softmax<<<cdiv((long)ETC * 8, T), T, 0, stream>>>(ei[g], logits, maxb, den);
            k_gat_aggregate<<<cdiv((long)ETC * 64, T), T, 0, stream>>>(xl, ei[g], logits, den, agg);
            if (L == 0) {
                k_bn_stats<<<DIMC, 256, 0, stream>>>(agg, meanb, rstdb);
                k_bn_elu_cvt<<<cdiv((long)NN * DIMC, T), T, 0, stream>>>(agg, bn_gamma, bn_beta,
                                                                         meanb, rstdb, xbm[g]);
            } else {
                k_cvt_bf16<<<cdiv((long)NN * DIMC, T), T, 0, stream>>>(agg, xfin[g], (long)NN * DIMC);
            }
        }
    }

    // ---- post network: gather-GEMM + fused (GEMM -> relu -> dot w3) ----
    k_init_scalar<<<cdiv(EJC, T), T, 0, stream>>>(out, post_b3, EJC);
    dim3 gE(EJC / 16);
    k_gemm_gather<<<gE, bG, 0, stream>>>(xfin[0], xfin[1], jg_ei, jg_ei + EJC,
                                         WTp1, post_b1, h1post);
    k_gemm_post2<<<gE, bG, 0, stream>>>(h1post, WTp2, post_b2, post_W3, out);
}
